// VQVAEV1_10488310136940
// MI455X (gfx1250) — matrix operand order repaired and checked by reading, not compiled, not run on hardware
//
#include <hip/hip_runtime.h>
#include <math.h>

typedef __bf16 v16bf __attribute__((ext_vector_type(16)));
typedef float  v8f   __attribute__((ext_vector_type(8)));

__device__ __forceinline__ float sigmoidf_(float x){ return 1.f/(1.f+__expf(-x)); }

// ---- WMMA fragment loaders (bf16 16x16x32, wave32; contiguous K order shared by A and B) ----
// A (16x32, MxK): lane m = lane%16, half = lane/16.
// element e (0..15): K = k0 + 16*half + e, i.e. 16 consecutive K values
// [k0+16*half, +16), matching the B fragment order.
__device__ __forceinline__ v16bf load_a_f32row(const float* Ar, int k0, int half){
  const float* p = Ar + k0 + (half<<4);
  float4 u0 = *(const float4*)(p);
  float4 u1 = *(const float4*)(p+4);
  float4 u2 = *(const float4*)(p+8);
  float4 u3 = *(const float4*)(p+12);
  v16bf a;
  a[0]=(__bf16)u0.x; a[1]=(__bf16)u0.y; a[2]=(__bf16)u0.z; a[3]=(__bf16)u0.w;
  a[4]=(__bf16)u1.x; a[5]=(__bf16)u1.y; a[6]=(__bf16)u1.z; a[7]=(__bf16)u1.w;
  a[8]=(__bf16)u2.x; a[9]=(__bf16)u2.y; a[10]=(__bf16)u2.z; a[11]=(__bf16)u2.w;
  a[12]=(__bf16)u3.x; a[13]=(__bf16)u3.y; a[14]=(__bf16)u3.z; a[15]=(__bf16)u3.w;
  return a;
}

// ---- bf16 weight pre-pack ----
__global__ void cvt_bf16(const float* __restrict__ in, __bf16* __restrict__ out, int n){
  int i = blockIdx.x*256 + threadIdx.x;
  if (i < n) out[i] = (__bf16)in[i];
}

// ---- generic WMMA GEMM:  Y[m][n] = epi( sum_k A[m][k]*W[n][k] + add[n] ) ----
// EPI: 0 = bias, 1 = bias+relu, 2 = VQ score (-2*acc + add[n])
template<int EPI>
__global__ __launch_bounds__(256) void wmma_gemm(const float* __restrict__ A,
                                                 const __bf16* __restrict__ W,
                                                 const float* __restrict__ add,
                                                 float* __restrict__ Y,
                                                 int tiles_n, int Kd, int N)
{
  int wid  = (blockIdx.x<<3) + (threadIdx.x>>5);
  int tm   = wid / tiles_n, tn = wid % tiles_n;
  int lane = threadIdx.x & 31;
  int half = lane >> 4, sub = lane & 15;
  const float*  Arow = A + (size_t)(tm*16 + sub) * Kd;
  const __bf16* Wrow = W + (size_t)(tn*16 + sub) * Kd;
  v8f acc = {0.f,0.f,0.f,0.f,0.f,0.f,0.f,0.f};
  for (int k0 = 0; k0 < Kd; k0 += 32){
    v16bf a = load_a_f32row(Arow, k0, half);
    v16bf b = *(const v16bf*)(Wrow + k0 + (half<<4));   // 16 consecutive K, 32B aligned
    acc = __builtin_amdgcn_wmma_f32_16x16x32_bf16(false, a, false, b, (short)0, acc, false, false);
  }
  int ng = tn*16 + sub;
  float av = add[ng];
  float* Yc = Y + (size_t)(tm*16 + (half<<3)) * N + ng;
#pragma unroll
  for (int r = 0; r < 8; ++r){
    float v;
    if (EPI == 2) v = -2.f*acc[r] + av;
    else { v = acc[r] + av; if (EPI == 1) v = fmaxf(v, 0.f); }
    Yc[(size_t)r*N] = v;
  }
}

// ---- BiLSTM scan: one workgroup per direction, 16 waves compute the 16x512
// gate GEMM (32 N-tiles) per step with WMMA; state lives in LDS ----
__global__ __launch_bounds__(512) void enc_lstm_scan(const float* __restrict__ encx,
    const __bf16* __restrict__ Wih_f, const __bf16* __restrict__ Whh_f, const float* __restrict__ bias_f,
    const __bf16* __restrict__ Wih_b, const __bf16* __restrict__ Whh_b, const float* __restrict__ bias_b,
    float* __restrict__ hcat)
{
  const int dir = blockIdx.x;
  const __bf16* Wih = dir ? Wih_b : Wih_f;
  const __bf16* Whh = dir ? Whh_b : Whh_f;
  const float*  bia = dir ? bias_b : bias_f;
  const int colOff  = dir ? 128 : 0;

  __shared__ __align__(16) float sG[16*512];
  __shared__ __align__(16) float sH[16*128];
  __shared__ __align__(16) float sC[16*128];
  const int tid = threadIdx.x;
  for (int i = tid; i < 16*128; i += 512){ sH[i] = 0.f; sC[i] = 0.f; }
  __syncthreads();

  const int wave = tid >> 5, lane = tid & 31;
  const int half = lane >> 4, sub = lane & 15;
  const float* hrow = sH + sub*128 - 128;          // used for combined-K >= 128

  for (int step = 0; step < 256; ++step){
    int t = dir ? (255 - step) : step;
    const float* xrow = encx + ((size_t)sub*256 + t)*128;
#pragma unroll
    for (int tt = 0; tt < 2; ++tt){
      int tn = wave*2 + tt;
      const __bf16* WihRow = Wih + (size_t)(tn*16 + sub)*128;
      const __bf16* WhhRow = Whh + (size_t)(tn*16 + sub)*128;
      v8f acc = {0.f,0.f,0.f,0.f,0.f,0.f,0.f,0.f};
      for (int k0 = 0; k0 < 256; k0 += 32){
        const float*  Arow = (k0 < 128) ? xrow : hrow;
        const __bf16* Wr   = (k0 < 128) ? (WihRow + k0) : (WhhRow + k0 - 128);
        v16bf a = load_a_f32row(Arow, k0, half);
        v16bf b = *(const v16bf*)(Wr + (half<<4));
        acc = __builtin_amdgcn_wmma_f32_16x16x32_bf16(false, a, false, b, (short)0, acc, false, false);
      }
      int ng = tn*16 + sub;
      float bb = bia[ng];
#pragma unroll
      for (int r = 0; r < 8; ++r) sG[(r + (half<<3))*512 + ng] = acc[r] + bb;
    }
    __syncthreads();
    for (int i = tid; i < 2048; i += 512){
      int bI = i >> 7, j = i & 127;
      float gi = sG[bI*512 + j],       gf = sG[bI*512 + 128 + j];
      float gg = sG[bI*512 + 256 + j], go = sG[bI*512 + 384 + j];
      float c = sigmoidf_(gf)*sC[i] + sigmoidf_(gi)*tanhf(gg);
      float h = sigmoidf_(go)*tanhf(c);
      sC[i] = c; sH[i] = h;
      hcat[((size_t)bI*256 + t)*256 + colOff + j] = h;
    }
    __syncthreads();
  }
}

// ---- codebook squared norms ----
__global__ void vq_norms(const float* __restrict__ emb, float* __restrict__ e2){
  int k = blockIdx.x*256 + threadIdx.x;
  if (k < 512){
    const float* r = emb + (size_t)k*128;
    float s = 0.f;
    for (int j = 0; j < 128; j += 4){
      float4 v = *(const float4*)(r + j);
      s += v.x*v.x + v.y*v.y + v.z*v.z + v.w*v.w;
    }
    e2[k] = s;
  }
}

// ---- argmin over K=512 per row (one wave per row) + gather zq = emb[nearest] ----
__global__ __launch_bounds__(256) void vq_argmin(const float* __restrict__ dist,
                                                 const float* __restrict__ emb,
                                                 float* __restrict__ zq)
{
  int row  = blockIdx.x*8 + (threadIdx.x>>5);
  int lane = threadIdx.x & 31;
  const float* d = dist + (size_t)row*512;
  float best = 3.4e38f; int bi = 0x7fffffff;
  for (int k = lane; k < 512; k += 32){
    float v = d[k];
    if (v < best || (v == best && k < bi)){ best = v; bi = k; }
  }
#pragma unroll
  for (int m = 16; m; m >>= 1){
    float ov = __shfl_xor(best, m, 32);
    int   oi = __shfl_xor(bi,   m, 32);
    if (ov < best || (ov == best && oi < bi)){ best = ov; bi = oi; }
  }
  const float* e = emb + (size_t)bi*128;
  float* z = zq + (size_t)row*128;
  for (int j = lane; j < 128; j += 32) z[j] = e[j];
}

// ---- decoder: one workgroup per batch element (independent across b).
// Per-b matmuls are matvecs (M=1) -> VALU; softmax via wave shuffles. ----
__global__ __launch_bounds__(256) void decoder_scan(
    const float* __restrict__ kproj, const float* __restrict__ vproj,
    const float* __restrict__ in_mask,
    const float* __restrict__ W1,  const float* __restrict__ b1,
    const float* __restrict__ Wih, const float* __restrict__ Whh, const float* __restrict__ bg,
    const float* __restrict__ Wq,  const float* __restrict__ bq,
    const float* __restrict__ W2,  const float* __restrict__ b2,
    float* __restrict__ dec_out, float* __restrict__ attn_out)
{
  const int b = blockIdx.x, tid = threadIdx.x;
  const int lane = tid & 31, wid = tid >> 5;
  __shared__ float sH[128], sC[128], sTok[64], sX[128], sQ[128], sG[512], sS[256], sCtx[128], sRed[8];
  if (tid < 128){ sH[tid] = 0.f; sC[tid] = 0.f; }
  if (tid < 64) sTok[tid] = 0.f;
  __syncthreads();
  const float* kp = kproj + (size_t)b*256*128;
  const float* vp = vproj + (size_t)b*256*128;
  const float scale = 0.08838834764831845f;  // 1/sqrt(128)

  for (int t = 0; t < 256; ++t){
    // x = relu(W1 @ tok + b1)
    if (tid < 128){
      const float* wr = W1 + (size_t)tid*64;
      float s = b1[tid];
      for (int k = 0; k < 64; k += 4){
        float4 w = *(const float4*)(wr + k);
        s += w.x*sTok[k] + w.y*sTok[k+1] + w.z*sTok[k+2] + w.w*sTok[k+3];
      }
      sX[tid] = fmaxf(s, 0.f);
    }
    __syncthreads();
    // gates = [x|h] @ [Wih|Whh]^T + bg
    for (int n = tid; n < 512; n += 256){
      const float* wi = Wih + (size_t)n*128;
      const float* wh = Whh + (size_t)n*128;
      float s = bg[n];
      for (int k = 0; k < 128; k += 4){
        float4 a = *(const float4*)(wi + k);
        s += a.x*sX[k] + a.y*sX[k+1] + a.z*sX[k+2] + a.w*sX[k+3];
      }
      for (int k = 0; k < 128; k += 4){
        float4 a = *(const float4*)(wh + k);
        s += a.x*sH[k] + a.y*sH[k+1] + a.z*sH[k+2] + a.w*sH[k+3];
      }
      sG[n] = s;
    }
    __syncthreads();
    // LSTM cell update
    if (tid < 128){
      float c = sigmoidf_(sG[128+tid])*sC[tid] + sigmoidf_(sG[tid])*tanhf(sG[256+tid]);
      float h = sigmoidf_(sG[384+tid])*tanhf(c);
      sC[tid] = c; sH[tid] = h;
    }
    __syncthreads();
    // q = Wq @ h + bq
    if (tid < 128){
      const float* wr = Wq + (size_t)tid*128;
      float s = bq[tid];
      for (int k = 0; k < 128; k += 4){
        float4 w = *(const float4*)(wr + k);
        s += w.x*sH[k] + w.y*sH[k+1] + w.z*sH[k+2] + w.w*sH[k+3];
      }
      sQ[tid] = s;
    }
    __syncthreads();
    // scores (tid = l) + softmax
    {
      const float* kr = kp + (size_t)tid*128;
      float s = 0.f;
      for (int k = 0; k < 128; k += 4){
        float4 w = *(const float4*)(kr + k);
        s += w.x*sQ[k] + w.y*sQ[k+1] + w.z*sQ[k+2] + w.w*sQ[k+3];
      }
      s = s*scale + (in_mask[b*256 + tid] == 0.f ? -1e9f : 0.f);
      sS[tid] = s;
      float v = s;
#pragma unroll
      for (int m = 16; m; m >>= 1) v = fmaxf(v, __shfl_xor(v, m, 32));
      if (lane == 0) sRed[wid] = v;
    }
    __syncthreads();
    float mx = sRed[0];
#pragma unroll
    for (int i = 1; i < 8; ++i) mx = fmaxf(mx, sRed[i]);
    float ev = __expf(sS[tid] - mx);
    __syncthreads();
    {
      float v = ev;
#pragma unroll
      for (int m = 16; m; m >>= 1) v += __shfl_xor(v, m, 32);
      if (lane == 0) sRed[wid] = v;
    }
    __syncthreads();
    float tot = 0.f;
#pragma unroll
    for (int i = 0; i < 8; ++i) tot += sRed[i];
    float at = ev / tot;
    sS[tid] = at;
    attn_out[((size_t)b*256 + t)*256 + tid] = at;
    __syncthreads();
    // ctx = attn @ V
    if (tid < 128){
      float s = 0.f;
      for (int l = 0; l < 256; ++l) s += sS[l] * vp[(size_t)l*128 + tid];
      sCtx[tid] = s;
    }
    __syncthreads();
    // out = W2 @ ctx + b2  (next token)
    if (tid < 64){
      const float* wr = W2 + (size_t)tid*128;
      float s = b2[tid];
      for (int k = 0; k < 128; k += 4){
        float4 w = *(const float4*)(wr + k);
        s += w.x*sCtx[k] + w.y*sCtx[k+1] + w.z*sCtx[k+2] + w.w*sCtx[k+3];
      }
      sTok[tid] = s;
      dec_out[((size_t)b*256 + t)*64 + tid] = s;
    }
    __syncthreads();
  }
}

extern "C" void kernel_launch(void* const* d_in, const int* in_sizes, int n_in,
                              void* d_out, int out_size, void* d_ws, size_t ws_size,
                              hipStream_t stream) {
  (void)in_sizes; (void)n_in; (void)out_size; (void)ws_size;
  // inputs (setup_inputs order)
  const float* inputs  = (const float*)d_in[0];
  const float* in_mask = (const float*)d_in[2];
  const float* e1W = (const float*)d_in[3],  *e1b = (const float*)d_in[4];
  const float* Wihf= (const float*)d_in[5],  *Whhf= (const float*)d_in[6],  *bf_ = (const float*)d_in[7];
  const float* Wihb= (const float*)d_in[8],  *Whhb= (const float*)d_in[9],  *bb_ = (const float*)d_in[10];
  const float* e2W = (const float*)d_in[11], *e2b = (const float*)d_in[12];
  const float* emb = (const float*)d_in[13];
  const float* d1W = (const float*)d_in[14], *d1b = (const float*)d_in[15];
  const float* dWih= (const float*)d_in[16], *dWhh= (const float*)d_in[17], *dbg = (const float*)d_in[18];
  const float* Wq  = (const float*)d_in[19], *bq  = (const float*)d_in[20];
  const float* Wk  = (const float*)d_in[21], *bk  = (const float*)d_in[22];
  const float* Wv  = (const float*)d_in[23], *bv  = (const float*)d_in[24];
  const float* d2W = (const float*)d_in[25], *d2b = (const float*)d_in[26];

  // output regions: dec_out | attn | ze | zq
  float* out    = (float*)d_out;
  float* decout = out;
  float* attn   = out + 262144;
  float* ze     = out + 1310720;
  float* zq     = out + 1835008;

  // workspace: bf16 weights, then f32 intermediates
  __bf16* w1bf  = (__bf16*)d_ws;
  __bf16* wihf  = w1bf + 8192;
  __bf16* whhf  = wihf + 65536;
  __bf16* wihb  = whhf + 65536;
  __bf16* whhb  = wihb + 65536;
  __bf16* w2bf  = whhb + 65536;
  __bf16* embbf = w2bf + 32768;
  __bf16* wkbf  = embbf + 65536;
  __bf16* wvbf  = wkbf + 16384;
  float*  encx  = (float*)(wvbf + 16384);
  float*  hcat  = encx + 524288;
  float*  e2v   = hcat + 1048576;
  float*  dist  = e2v + 512;
  float*  kproj = dist + 2097152;
  float*  vproj = kproj + 524288;

  // 1) pack weights to bf16
  cvt_bf16<<<32,  256, 0, stream>>>(e1W,  w1bf,  8192);
  cvt_bf16<<<256, 256, 0, stream>>>(Wihf, wihf,  65536);
  cvt_bf16<<<256, 256, 0, stream>>>(Whhf, whhf,  65536);
  cvt_bf16<<<256, 256, 0, stream>>>(Wihb, wihb,  65536);
  cvt_bf16<<<256, 256, 0, stream>>>(Whhb, whhb,  65536);
  cvt_bf16<<<128, 256, 0, stream>>>(e2W,  w2bf,  32768);
  cvt_bf16<<<256, 256, 0, stream>>>(emb,  embbf, 65536);
  cvt_bf16<<<64,  256, 0, stream>>>(Wk,   wkbf,  16384);
  cvt_bf16<<<64,  256, 0, stream>>>(Wv,   wvbf,  16384);

  // 2) encoder lin1 + relu : (4096x64)@(64x128) -> encx
  wmma_gemm<1><<<256, 256, 0, stream>>>(inputs, w1bf, e1b, encx, 8, 64, 128);
  // 3) BiLSTM scan (2 directions) -> hcat (4096x256)
  enc_lstm_scan<<<2, 512, 0, stream>>>(encx, wihf, whhf, bf_, wihb, whhb, bb_, hcat);
  // 4) encoder lin2 : (4096x256)@(256x128) -> ze (d_out)
  wmma_gemm<0><<<256, 256, 0, stream>>>(hcat, w2bf, e2b, ze, 8, 256, 128);
  // 5) VQ: scores = -2*ze@emb^T + ||emb||^2 ; argmin ; gather zq
  vq_norms<<<2, 256, 0, stream>>>(emb, e2v);
  wmma_gemm<2><<<1024, 256, 0, stream>>>(ze, embbf, e2v, dist, 32, 128, 512);
  vq_argmin<<<512, 256, 0, stream>>>(dist, emb, zq);
  // 6) attention K/V projections from dec_in (== zq in forward)
  wmma_gemm<0><<<256, 256, 0, stream>>>(zq, wkbf, bk, kproj, 8, 128, 128);
  wmma_gemm<0><<<256, 256, 0, stream>>>(zq, wvbf, bv, vproj, 8, 128, 128);
  // 7) decoder: one WG per batch element
  decoder_scan<<<16, 256, 0, stream>>>(kproj, vproj, in_mask,
                                       d1W, d1b, dWih, dWhh, dbg,
                                       Wq, bq, d2W, d2b, decout, attn);
}
